// HighResPointMambaUNet_65163243815255
// MI455X (gfx1250) — compile-verified, matrix operand order checked by reading, not run on hardware
//
#include <hip/hip_runtime.h>

// ---------------------------------------------------------------------------
// CDNA5 / gfx1250 implementation of HighResPointMambaUNet forward.
// Heavy GEMMs run on v_wmma_f32_16x16x32_f16 (wave32, 16x16x32 tiles).
// All GEMM operands are zero-padded (K->mult of 64, M/N->mult of 64) so the
// inner K-loop has no bounds checks: pure load + wmma.
// ---------------------------------------------------------------------------

typedef __attribute__((ext_vector_type(16))) _Float16 v16h;
typedef __attribute__((ext_vector_type(8)))  _Float16 v8h;
typedef __attribute__((ext_vector_type(8)))  float    v8f;

// 16-bit A/B fragment load per CDNA5 ISA layout:
// lanes 0-15 hold K = [0..7] (VGPR0-3) and [16..23] (VGPR4-7),
// lanes 16-31 hold K = [8..15] and [24..31]. Two aligned 16B chunks.
__device__ __forceinline__ v16h load_frag(const _Float16* rowp, int k0, int half) {
  const v8h lo = *(const v8h*)(rowp + k0 + half * 8);
  const v8h hi = *(const v8h*)(rowp + k0 + 16 + half * 8);
  v16h r;
#pragma unroll
  for (int e = 0; e < 8; ++e) { r[e] = lo[e]; r[e + 8] = hi[e]; }
  return r;
}

__device__ __forceinline__ float siluf(float v) { return v / (1.f + __expf(-v)); }

// ---------------------------------------------------------------------------
// GEMM: C[M,N] = act(A[Mp,Kp] * Bt[Np,Kp]^T + bias[N])
// A, Bt f16 row-major, zero-padded: Kp % 64 == 0, rows padded to grid extent.
// act: 0 = none, 1 = gelu(erf), 2 = softplus
// Block = 128 threads (4 waves); each wave computes a 32x32 tile (2x2 WMMA),
// K unrolled by 2 -> 8 v_wmma per iteration, no control flow in the loop.
// ---------------------------------------------------------------------------
__global__ void k_wmma_gemm(const _Float16* __restrict__ A,
                            const _Float16* __restrict__ Bt,
                            const float* __restrict__ bias,
                            float* __restrict__ C,
                            int M, int N, int Kp, int act) {
  int lane  = threadIdx.x & 31;
  int wave  = threadIdx.x >> 5;
  int waveM = wave >> 1, waveN = wave & 1;
  int m0 = blockIdx.y * 64 + waveM * 32;
  int n0 = blockIdx.x * 64 + waveN * 32;
  int half = lane >> 4;
  int l16  = lane & 15;

  const _Float16* arow[2];
  const _Float16* brow[2];
#pragma unroll
  for (int i = 0; i < 2; ++i) arow[i] = A  + (size_t)(m0 + i * 16 + l16) * Kp;
#pragma unroll
  for (int j = 0; j < 2; ++j) brow[j] = Bt + (size_t)(n0 + j * 16 + l16) * Kp;

  v8f acc[2][2];
#pragma unroll
  for (int i = 0; i < 2; ++i)
#pragma unroll
    for (int j = 0; j < 2; ++j)
#pragma unroll
      for (int r = 0; r < 8; ++r) acc[i][j][r] = 0.f;

  for (int k0 = 0; k0 < Kp; k0 += 64) {
    v16h a0[2], a1[2], b0[2], b1[2];
#pragma unroll
    for (int i = 0; i < 2; ++i) {
      a0[i] = load_frag(arow[i], k0, half);
      a1[i] = load_frag(arow[i], k0 + 32, half);
    }
#pragma unroll
    for (int j = 0; j < 2; ++j) {
      b0[j] = load_frag(brow[j], k0, half);
      b1[j] = load_frag(brow[j], k0 + 32, half);
    }
    __builtin_prefetch(arow[0] + k0 + 128, 0, 1);   // global_prefetch_b8
    __builtin_prefetch(brow[0] + k0 + 128, 0, 1);
#pragma unroll
    for (int i = 0; i < 2; ++i)
#pragma unroll
      for (int j = 0; j < 2; ++j)
        acc[i][j] = __builtin_amdgcn_wmma_f32_16x16x32_f16(
            false, a0[i], false, b0[j], (short)0, acc[i][j], false, false);
#pragma unroll
    for (int i = 0; i < 2; ++i)
#pragma unroll
      for (int j = 0; j < 2; ++j)
        acc[i][j] = __builtin_amdgcn_wmma_f32_16x16x32_f16(
            false, a1[i], false, b1[j], (short)0, acc[i][j], false, false);
  }

#pragma unroll
  for (int i = 0; i < 2; ++i) {
#pragma unroll
    for (int j = 0; j < 2; ++j) {
      int col = n0 + j * 16 + l16;
      if (col >= N) continue;
      float bv = bias ? bias[col] : 0.f;
#pragma unroll
      for (int r = 0; r < 8; ++r) {
        // C/D layout: VGPR r -> M = r (lanes 0-15), M = r+8 (lanes 16-31)
        int row = m0 + i * 16 + half * 8 + r;
        if (row < M) {
          float v = acc[i][j][r] + bv;
          if (act == 1)      v = 0.5f * v * (1.f + erff(v * 0.70710678118f));
          else if (act == 2) v = (v > 20.f) ? v : log1pf(__expf(v));
          C[(size_t)row * N + col] = v;
        }
      }
    }
  }
}

// W (K x N, f32, row-major) -> Wt (Np x Kp, f16, row-major, zero pad)
__global__ void k_transpose_w(const float* __restrict__ W, _Float16* __restrict__ Wt,
                              int K, int N, int Kp, int Np) {
  int idx = blockIdx.x * blockDim.x + threadIdx.x;
  if (idx >= Np * Kp) return;
  int n = idx / Kp, k = idx % Kp;
  Wt[idx] = (k < K && n < N) ? (_Float16)W[(size_t)k * N + n] : (_Float16)0.f;
}

// f32 (strided, optional offset, optional silu) -> f16 row-major [Mp, Kp]
// rows >= Mvalid and cols >= C are zero-filled.
__global__ void k_convert_pad(const float* __restrict__ src, int srcStride, int srcOff,
                              int C, _Float16* __restrict__ dst, int Kp, int Mp,
                              int Mvalid, int act) {
  int idx = blockIdx.x * blockDim.x + threadIdx.x;
  if (idx >= Mp * Kp) return;
  int i = idx / Kp, c = idx % Kp;
  float v = 0.f;
  if (c < C && i < Mvalid) {
    v = src[(size_t)i * srcStride + srcOff + c];
    if (act == 1) v = siluf(v);
  }
  dst[idx] = (_Float16)v;
}

// LayerNorm + AdaLN modulation -> f16
__global__ void k_ln_mod_f16(const float* __restrict__ x, int d,
                             const float* __restrict__ mod, int shOff, int scOff,
                             _Float16* __restrict__ out, int Kp, float eps) {
  __shared__ float red[256];
  int row = blockIdx.x, tid = threadIdx.x;
  const float* xr = x + (size_t)row * d;
  float s = 0.f;
  for (int c = tid; c < d; c += 256) s += xr[c];
  red[tid] = s; __syncthreads();
  for (int w = 128; w > 0; w >>= 1) { if (tid < w) red[tid] += red[tid + w]; __syncthreads(); }
  float mean = red[0] / d; __syncthreads();
  float v = 0.f;
  for (int c = tid; c < d; c += 256) { float t = xr[c] - mean; v += t * t; }
  red[tid] = v; __syncthreads();
  for (int w = 128; w > 0; w >>= 1) { if (tid < w) red[tid] += red[tid + w]; __syncthreads(); }
  float inv = rsqrtf(red[0] / d + eps);
  for (int c = tid; c < Kp; c += 256) {
    float o = 0.f;
    if (c < d) {
      float n = (xr[c] - mean) * inv;
      o = n * (1.f + mod[scOff + c]) + mod[shOff + c];
    }
    out[(size_t)row * Kp + c] = (_Float16)o;
  }
}

// Final LayerNorm * g + beta -> f32
__global__ void k_ln_affine(const float* __restrict__ x, int d,
                            const float* __restrict__ g, const float* __restrict__ beta,
                            float* __restrict__ out, float eps) {
  __shared__ float red[256];
  int row = blockIdx.x, tid = threadIdx.x;
  const float* xr = x + (size_t)row * d;
  float s = 0.f;
  for (int c = tid; c < d; c += 256) s += xr[c];
  red[tid] = s; __syncthreads();
  for (int w = 128; w > 0; w >>= 1) { if (tid < w) red[tid] += red[tid + w]; __syncthreads(); }
  float mean = red[0] / d; __syncthreads();
  float v = 0.f;
  for (int c = tid; c < d; c += 256) { float t = xr[c] - mean; v += t * t; }
  red[tid] = v; __syncthreads();
  for (int w = 128; w > 0; w >>= 1) { if (tid < w) red[tid] += red[tid + w]; __syncthreads(); }
  float inv = rsqrtf(red[0] / d + eps);
  for (int c = tid; c < d; c += 256)
    out[(size_t)row * d + c] = (xr[c] - mean) * inv * g[c] + beta[c];
}

// causal depthwise conv (D_CONV=4) + bias + silu -> f32 and f16 copies
__global__ void k_conv_silu(const float* __restrict__ xz, int xzStride,
                            const float* __restrict__ w, const float* __restrict__ b,
                            float* __restrict__ xc32, _Float16* __restrict__ xc16,
                            int Kp, int L, int di) {
  int idx = blockIdx.x * blockDim.x + threadIdx.x;
  if (idx >= L * di) return;
  int t = idx / di, c = idx % di;
  float acc = b[c];
#pragma unroll
  for (int k = 0; k < 4; ++k) {
    int tt = t - 3 + k;
    if (tt >= 0) acc += xz[(size_t)tt * xzStride + c] * w[c * 4 + k];
  }
  float v = siluf(acc);
  xc32[(size_t)t * di + c] = v;
  xc16[(size_t)t * Kp + c] = (_Float16)v;
}

// Selective scan: lane = (channel, state); 16-lane shuffle reduction per step.
__global__ void k_scan(const float* __restrict__ dt, const float* __restrict__ xc,
                       const float* __restrict__ Bm, const float* __restrict__ Cm,
                       int bStride, const float* __restrict__ A_log,
                       const float* __restrict__ Dp, float* __restrict__ y,
                       int L, int di) {
  int s  = threadIdx.x & 15;
  int ch = threadIdx.x >> 4;
  int d  = blockIdx.x * 16 + ch;
  if (d >= di) return;
  float A  = -__expf(A_log[d * 16 + s]);
  float Dv = Dp[d];
  float h = 0.f;
  for (int t = 0; t < L; ++t) {
    float dtv = dt[(size_t)t * di + d];
    float xv  = xc[(size_t)t * di + d];
    float Bv  = Bm[(size_t)t * bStride + s];
    float Cv  = Cm[(size_t)t * bStride + s];
    h = __expf(dtv * A) * h + dtv * xv * Bv;
    float p = h * Cv;
    p += __shfl_xor(p, 1, 32);
    p += __shfl_xor(p, 2, 32);
    p += __shfl_xor(p, 4, 32);
    p += __shfl_xor(p, 8, 32);
    if (s == 0) y[(size_t)t * di + d] = p + xv * Dv;
  }
}

__global__ void k_gate(const float* __restrict__ y, const float* __restrict__ xz,
                       int stride, int zOff, _Float16* __restrict__ out, int L, int di) {
  int idx = blockIdx.x * blockDim.x + threadIdx.x;
  if (idx >= L * di) return;
  int t = idx / di, c = idx % di;
  float z = xz[(size_t)t * stride + zOff + c];
  out[idx] = (_Float16)(y[(size_t)t * di + c] * siluf(z));
}

__global__ void k_flip16(const _Float16* __restrict__ src, _Float16* __restrict__ dst,
                         int L, int d) {
  int idx = blockIdx.x * blockDim.x + threadIdx.x;
  if (idx >= L * d) return;
  int t = idx / d, c = idx % d;
  dst[(size_t)(L - 1 - t) * d + c] = src[idx];
}

__global__ void k_add(float* __restrict__ a, const float* __restrict__ b, int n) {
  int idx = blockIdx.x * blockDim.x + threadIdx.x;
  if (idx < n) a[idx] += b[idx];
}

__global__ void k_add_flipped(float* __restrict__ a, const float* __restrict__ b,
                              int L, int d) {
  int idx = blockIdx.x * blockDim.x + threadIdx.x;
  if (idx >= L * d) return;
  int t = idx / d, c = idx % d;
  a[idx] += b[(size_t)(L - 1 - t) * d + c];
}

__global__ void k_gather_f16(const long long* __restrict__ idxs,
                             const float* __restrict__ src, int C,
                             _Float16* __restrict__ dst, int Kp, int M) {
  int idx = blockIdx.x * blockDim.x + threadIdx.x;
  if (idx >= M * Kp) return;
  int i = idx / Kp, c = idx % Kp;
  dst[idx] = (c < C) ? (_Float16)src[(size_t)idxs[i] * C + c] : (_Float16)0.f;
}

__global__ void k_upsample_f16(const float* __restrict__ src, int Ls, int C,
                               _Float16* __restrict__ dst, int Lt) {
  int idx = blockIdx.x * blockDim.x + threadIdx.x;
  if (idx >= Lt * C) return;
  int i = idx / C, c = idx % C;
  size_t row = ((size_t)i * Ls) / Lt;
  dst[idx] = (_Float16)src[row * C + c];
}

__global__ void k_concat2_f16(const float* __restrict__ a, int C1,
                              const float* __restrict__ b, int C2,
                              _Float16* __restrict__ dst, int L) {
  int Kp = C1 + C2;
  int idx = blockIdx.x * blockDim.x + threadIdx.x;
  if (idx >= L * Kp) return;
  int i = idx / Kp, c = idx % Kp;
  float v = (c < C1) ? a[(size_t)i * C1 + c] : b[(size_t)i * C2 + (c - C1)];
  dst[idx] = (_Float16)v;
}

// x_input row = [x | point_feat | xyz | 0-pad], gathered by sorted index, Kp=576
__global__ void k_build_xin(const int* __restrict__ sidx, const float* __restrict__ x,
                            const float* __restrict__ pf, const float* __restrict__ xyz,
                            _Float16* __restrict__ out, int N) {
  int idx = blockIdx.x * blockDim.x + threadIdx.x;
  if (idx >= N * 576) return;
  int i = idx / 576, c = idx % 576;
  long long sr = sidx[i];
  float v = 0.f;
  if (c == 0)        v = x[sr];
  else if (c < 513)  v = pf[(size_t)sr * 512 + (c - 1)];
  else if (c < 516)  v = xyz[(size_t)sr * 3 + (c - 513)];
  out[idx] = (_Float16)v;
}

__global__ void k_scatter_out(const int* __restrict__ sidx, const float* __restrict__ src,
                              float* __restrict__ dst, int N, int d) {
  int idx = blockIdx.x * blockDim.x + threadIdx.x;
  if (idx >= N * d) return;
  int i = idx / d, c = idx % d;
  dst[(size_t)sidx[i] * d + c] = src[idx];
}

__global__ void k_concat_cond(const float* __restrict__ lang, const float* __restrict__ tim,
                              float* __restrict__ dst) {
  int c = blockIdx.x * blockDim.x + threadIdx.x;
  if (c < 640) dst[c] = (c < 512) ? lang[c] : tim[c - 512];
}

__global__ void k_copy_pad_f32(float* __restrict__ dst, int n,
                               const float* __restrict__ src, int m) {
  int i = blockIdx.x * blockDim.x + threadIdx.x;
  if (i < n) dst[i] = (i < m) ? src[i] : 0.f;
}

__global__ void k_minmax(const float* __restrict__ xyz, int N,
                         float* __restrict__ mn, float* __restrict__ mx) {
  __shared__ float smn[3][256], smx[3][256];
  int tid = threadIdx.x;
  float lmn[3] = {1e30f, 1e30f, 1e30f}, lmx[3] = {-1e30f, -1e30f, -1e30f};
  for (int i = tid; i < N; i += 256) {
#pragma unroll
    for (int a = 0; a < 3; ++a) {
      float v = xyz[(size_t)i * 3 + a];
      lmn[a] = fminf(lmn[a], v);
      lmx[a] = fmaxf(lmx[a], v);
    }
  }
#pragma unroll
  for (int a = 0; a < 3; ++a) { smn[a][tid] = lmn[a]; smx[a][tid] = lmx[a]; }
  __syncthreads();
  for (int w = 128; w > 0; w >>= 1) {
    if (tid < w)
#pragma unroll
      for (int a = 0; a < 3; ++a) {
        smn[a][tid] = fminf(smn[a][tid], smn[a][tid + w]);
        smx[a][tid] = fmaxf(smx[a][tid], smx[a][tid + w]);
      }
    __syncthreads();
  }
  if (tid < 3) { mn[tid] = smn[tid][0]; mx[tid] = smx[tid][0]; }
}

__global__ void k_hilbert(const float* __restrict__ xyz, const float* __restrict__ mn,
                          const float* __restrict__ mx, unsigned long long* __restrict__ codes,
                          int N) {
  int i = blockIdx.x * blockDim.x + threadIdx.x;
  if (i >= N) return;
  unsigned X[3];
#pragma unroll
  for (int a = 0; a < 3; ++a) {
    float nrm = (xyz[(size_t)i * 3 + a] - mn[a]) / (mx[a] - mn[a] + 1e-6f);
    float sc = nrm * 65535.f;
    sc = fminf(fmaxf(sc, 0.f), 65535.f);
    X[a] = (unsigned)sc;
  }
  const unsigned M = 1u << 15;
  for (unsigned Q = M; Q > 1; Q >>= 1) {
    unsigned P = Q - 1;
    if (X[0] & Q) X[0] ^= P;
    for (int a = 1; a < 3; ++a) {
      if (X[a] & Q) { X[0] ^= P; }
      else { unsigned t = (X[0] ^ X[a]) & P; X[0] ^= t; X[a] ^= t; }
    }
  }
  X[1] ^= X[0];
  X[2] ^= X[1];
  unsigned t2 = 0;
  for (unsigned Q = M; Q > 1; Q >>= 1)
    if (X[2] & Q) t2 ^= (Q - 1);
#pragma unroll
  for (int a = 0; a < 3; ++a) X[a] ^= t2;
  unsigned long long code = 0ull;
  for (int b = 15; b >= 0; --b)
#pragma unroll
    for (int a = 0; a < 3; ++a)
      code = (code << 1) | ((X[a] >> b) & 1u);
  codes[i] = code;
}

__global__ void k_iota(int* __restrict__ v, int n) {
  int i = blockIdx.x * blockDim.x + threadIdx.x;
  if (i < n) v[i] = i;
}

__global__ void k_bitonic(unsigned long long* __restrict__ key, int* __restrict__ val,
                          int j, int k, int n) {
  int i = blockIdx.x * blockDim.x + threadIdx.x;
  if (i >= n) return;
  int ixj = i ^ j;
  if (ixj > i && ixj < n) {
    bool up = ((i & k) == 0);
    unsigned long long ki = key[i], kj = key[ixj];
    bool sw = up ? (ki > kj) : (ki < kj);
    if (sw) {
      key[i] = kj; key[ixj] = ki;
      int t = val[i]; val[i] = val[ixj]; val[ixj] = t;
    }
  }
}

// ---------------------------------------------------------------------------
// Host orchestration
// ---------------------------------------------------------------------------
struct Bump {
  char* p; size_t off;
  void* take(size_t bytes) {
    void* r = p + off;
    off = (off + bytes + 255) & ~(size_t)255;
    return r;
  }
};

static inline int cdiv(int a, int b) { return (a + b - 1) / b; }
static inline int rup64(int a) { return (a + 63) & ~63; }

extern "C" void kernel_launch(void* const* d_in, const int* in_sizes, int n_in,
                              void* d_out, int out_size, void* d_ws, size_t ws_size,
                              hipStream_t stream) {
  (void)in_sizes; (void)n_in; (void)out_size; (void)ws_size;
  const int NPTS = 8192, D0 = 128;

  const float*     xin  = (const float*)d_in[0];
  const float*     pf   = (const float*)d_in[1];
  const float*     lang = (const float*)d_in[2];
  const float*     tim  = (const float*)d_in[3];
  const float*     xyz  = (const float*)d_in[4];
  const long long* id1  = (const long long*)d_in[5];
  const long long* id2  = (const long long*)d_in[6];
  auto PR = [&](int i) -> const float* { return (const float*)d_in[7 + i]; };

  // param-relative indices (flattened in dict insertion order)
  const int P_condW = 0, P_condB = 1, P_inW = 2, P_inB = 3;
  const int P_ENC1 = 4, P_down1W = 28, P_down1B = 29;
  const int P_ENC2 = 30, P_down2W = 54, P_down2B = 55;
  const int P_MID = 56, P_up2W = 80, P_up2B = 81, P_dec2pW = 82, P_dec2pB = 83;
  const int P_DEC2 = 84, P_up1W = 108, P_up1B = 109, P_dec1pW = 110, P_dec1pB = 111;
  const int P_DEC1 = 112, P_outG = 136, P_outBeta = 137;
  // within a block: ada_W=+0 ada_b=+1 fwd=+2..10 bwd=+11..19 W1=+20 b1=+21 W2=+22 b2=+23
  // within a mamba: W_in=0 conv_w=1 conv_b=2 W_x=3 W_dt=4 b_dt=5 A_log=6 D=7 W_out=8

  Bump ws{(char*)d_ws, 0};

  auto gemm = [&](const _Float16* A, int M, int Kp, const float* W, int K, int Nn,
                  const float* bias, float* C, int act, Bump& s) {
    int Np = rup64(Nn);
    _Float16* Wt = (_Float16*)s.take((size_t)Np * Kp * sizeof(_Float16));
    int tot = Np * Kp;
    k_transpose_w<<<cdiv(tot, 256), 256, 0, stream>>>(W, Wt, K, Nn, Kp, Np);
    dim3 g(cdiv(Nn, 64), cdiv(M, 64));
    k_wmma_gemm<<<g, 128, 0, stream>>>(A, Wt, bias, C, M, Nn, Kp, act);
  };

  auto mamba = [&](const _Float16* X16, int L, int d, int pm, float* out, Bump s) {
    int di = 2 * d, dr = (d + 15) / 16, nx = dr + 32;
    float* xz = (float*)s.take((size_t)L * 2 * di * sizeof(float));
    gemm(X16, L, d, PR(pm + 0), d, 2 * di, nullptr, xz, 0, s);
    float*     xc32 = (float*)s.take((size_t)L * di * sizeof(float));
    _Float16*  xc16 = (_Float16*)s.take((size_t)L * di * sizeof(_Float16));
    k_conv_silu<<<cdiv(L * di, 256), 256, 0, stream>>>(xz, 2 * di, PR(pm + 1), PR(pm + 2),
                                                       xc32, xc16, di, L, di);
    float* xdbl = (float*)s.take((size_t)L * nx * sizeof(float));
    gemm(xc16, L, di, PR(pm + 3), di, nx, nullptr, xdbl, 0, s);
    _Float16* dtA = (_Float16*)s.take((size_t)L * 64 * sizeof(_Float16));
    k_convert_pad<<<cdiv(L * 64, 256), 256, 0, stream>>>(xdbl, nx, 0, dr, dtA, 64, L, L, 0);
    float* dt = (float*)s.take((size_t)L * di * sizeof(float));
    gemm(dtA, L, 64, PR(pm + 4), dr, di, PR(pm + 5), dt, 2 /*softplus*/, s);
    float* y = (float*)s.take((size_t)L * di * sizeof(float));
    k_scan<<<di / 16, 256, 0, stream>>>(dt, xc32, xdbl + dr, xdbl + dr + 16, nx,
                                        PR(pm + 6), PR(pm + 7), y, L, di);
    _Float16* yg = (_Float16*)s.take((size_t)L * di * sizeof(_Float16));
    k_gate<<<cdiv(L * di, 256), 256, 0, stream>>>(y, xz, 2 * di, di, yg, L, di);
    gemm(yg, L, di, PR(pm + 8), di, d, nullptr, out, 0, s);
  };

  auto bimamba = [&](float* x, int L, int d, const float* cond, int pb, Bump s) {
    // modulation: mod = silu(cond) @ ada_W + ada_b  (M=1 padded to 64 rows)
    _Float16* c16 = (_Float16*)s.take((size_t)64 * d * sizeof(_Float16));
    k_convert_pad<<<cdiv(64 * d, 256), 256, 0, stream>>>(cond, d, 0, d, c16, d, 64, 1, 1);
    float* mod = (float*)s.take((size_t)4 * d * sizeof(float));
    gemm(c16, 1, d, PR(pb + 0), d, 4 * d, PR(pb + 1), mod, 0, s);
    // MSA branch
    _Float16* hln  = (_Float16*)s.take((size_t)L * d * sizeof(_Float16));
    _Float16* hlnf = (_Float16*)s.take((size_t)L * d * sizeof(_Float16));
    float* hsum = (float*)s.take((size_t)L * d * sizeof(float));
    float* htmp = (float*)s.take((size_t)L * d * sizeof(float));
    k_ln_mod_f16<<<L, 256, 0, stream>>>(x, d, mod, 0, d, hln, d, 1e-6f);
    k_flip16<<<cdiv(L * d, 256), 256, 0, stream>>>(hln, hlnf, L, d);
    mamba(hln, L, d, pb + 2, hsum, s);
    mamba(hlnf, L, d, pb + 11, htmp, s);
    k_add_flipped<<<cdiv(L * d, 256), 256, 0, stream>>>(hsum, htmp, L, d);
    k_add<<<cdiv(L * d, 256), 256, 0, stream>>>(x, hsum, L * d);
    // MLP branch
    k_ln_mod_f16<<<L, 256, 0, stream>>>(x, d, mod, 2 * d, 3 * d, hln, d, 1e-6f);
    float* m1 = (float*)s.take((size_t)L * 2 * d * sizeof(float));
    gemm(hln, L, d, PR(pb + 20), d, 2 * d, PR(pb + 21), m1, 1 /*gelu*/, s);
    _Float16* m116 = (_Float16*)s.take((size_t)L * 2 * d * sizeof(_Float16));
    k_convert_pad<<<cdiv(L * 2 * d, 256), 256, 0, stream>>>(m1, 2 * d, 0, 2 * d, m116,
                                                            2 * d, L, L, 0);
    gemm(m116, L, 2 * d, PR(pb + 22), 2 * d, d, PR(pb + 23), htmp, 0, s);
    k_add<<<cdiv(L * d, 256), 256, 0, stream>>>(x, htmp, L * d);
  };

  // ---- Hilbert encode + sort -------------------------------------------------
  float* mnmx = (float*)ws.take(8 * sizeof(float));
  unsigned long long* codes = (unsigned long long*)ws.take((size_t)NPTS * 8);
  int* sidx = (int*)ws.take((size_t)NPTS * 4);
  k_minmax<<<1, 256, 0, stream>>>(xyz, NPTS, mnmx, mnmx + 4);
  k_hilbert<<<cdiv(NPTS, 256), 256, 0, stream>>>(xyz, mnmx, mnmx + 4, codes, NPTS);
  k_iota<<<cdiv(NPTS, 256), 256, 0, stream>>>(sidx, NPTS);
  for (int k = 2; k <= NPTS; k <<= 1)
    for (int j = k >> 1; j > 0; j >>= 1)
      k_bitonic<<<cdiv(NPTS, 256), 256, 0, stream>>>(codes, sidx, j, k, NPTS);

  // ---- Input embedding (K=516 padded to 576) ---------------------------------
  _Float16* xin16 = (_Float16*)ws.take((size_t)NPTS * 576 * sizeof(_Float16));
  k_build_xin<<<cdiv(NPTS * 576, 256), 256, 0, stream>>>(sidx, xin, pf, xyz, xin16, NPTS);

  // cond vectors
  float* c640 = (float*)ws.take(640 * sizeof(float));
  _Float16* c64016 = (_Float16*)ws.take((size_t)64 * 640 * sizeof(_Float16));
  float* c128 = (float*)ws.take(128 * sizeof(float));
  float* c256 = (float*)ws.take(256 * sizeof(float));
  float* c512 = (float*)ws.take(512 * sizeof(float));
  k_concat_cond<<<cdiv(640, 256), 256, 0, stream>>>(lang, tim, c640);
  k_convert_pad<<<cdiv(64 * 640, 256), 256, 0, stream>>>(c640, 640, 0, 640, c64016,
                                                         640, 64, 1, 0);
  { Bump s = ws; gemm(c64016, 1, 640, PR(P_condW), 640, 128, PR(P_condB), c128, 0, s); }
  k_copy_pad_f32<<<1, 256, 0, stream>>>(c256, 256, c128, 128);
  k_copy_pad_f32<<<cdiv(512, 256), 256, 0, stream>>>(c512, 512, c128, 128);

  // x0 = x_input @ in_W + in_b
  float* x0 = (float*)ws.take((size_t)NPTS * D0 * sizeof(float));
  { Bump s = ws; gemm(xin16, NPTS, 576, PR(P_inW), 516, 128, PR(P_inB), x0, 0, s); }

  // ---- Encoder ---------------------------------------------------------------
  { Bump s = ws; bimamba(x0, NPTS, 128, c128, P_ENC1, s); }           // x1_skip = x0
  const int L1 = NPTS / 4, L2 = NPTS / 16;
  _Float16* g1 = (_Float16*)ws.take((size_t)L1 * 128 * sizeof(_Float16));
  k_gather_f16<<<cdiv(L1 * 128, 256), 256, 0, stream>>>(id1, x0, 128, g1, 128, L1);
  float* x1 = (float*)ws.take((size_t)L1 * 256 * sizeof(float));
  { Bump s = ws; gemm(g1, L1, 128, PR(P_down1W), 128, 256, PR(P_down1B), x1, 0, s); }

  { Bump s = ws; bimamba(x1, L1, 256, c256, P_ENC2, s); }             // x2_skip = x1
  _Float16* g2 = (_Float16*)ws.take((size_t)L2 * 256 * sizeof(_Float16));
  k_gather_f16<<<cdiv(L2 * 256, 256), 256, 0, stream>>>(id2, x1, 256, g2, 256, L2);
  float* x2 = (float*)ws.take((size_t)L2 * 512 * sizeof(float));
  { Bump s = ws; gemm(g2, L2, 256, PR(P_down2W), 256, 512, PR(P_down2B), x2, 0, s); }

  // ---- Mid -------------------------------------------------------------------
  { Bump s = ws; bimamba(x2, L2, 512, c512, P_MID, s); }              // x_mid = x2

  // ---- Decoder level 2 -------------------------------------------------------
  _Float16* up2in = (_Float16*)ws.take((size_t)L1 * 512 * sizeof(_Float16));
  k_upsample_f16<<<cdiv(L1 * 512, 256), 256, 0, stream>>>(x2, L2, 512, up2in, L1);
  float* up2 = (float*)ws.take((size_t)L1 * 256 * sizeof(float));
  { Bump s = ws; gemm(up2in, L1, 512, PR(P_up2W), 512, 256, PR(P_up2B), up2, 0, s); }
  _Float16* cat2 = (_Float16*)ws.take((size_t)L1 * 512 * sizeof(_Float16));
  k_concat2_f16<<<cdiv(L1 * 512, 256), 256, 0, stream>>>(up2, 256, x1, 256, cat2, L1);
  float* xu2 = (float*)ws.take((size_t)L1 * 256 * sizeof(float));
  { Bump s = ws; gemm(cat2, L1, 512, PR(P_dec2pW), 512, 256, PR(P_dec2pB), xu2, 0, s); }
  { Bump s = ws; bimamba(xu2, L1, 256, c256, P_DEC2, s); }

  // ---- Decoder level 1 -------------------------------------------------------
  _Float16* up1in = (_Float16*)ws.take((size_t)NPTS * 256 * sizeof(_Float16));
  k_upsample_f16<<<cdiv(NPTS * 256, 256), 256, 0, stream>>>(xu2, L1, 256, up1in, NPTS);
  float* up1 = (float*)ws.take((size_t)NPTS * 128 * sizeof(float));
  { Bump s = ws; gemm(up1in, NPTS, 256, PR(P_up1W), 256, 128, PR(P_up1B), up1, 0, s); }
  _Float16* cat1 = (_Float16*)ws.take((size_t)NPTS * 256 * sizeof(_Float16));
  k_concat2_f16<<<cdiv(NPTS * 256, 256), 256, 0, stream>>>(up1, 128, x0, 128, cat1, NPTS);
  float* xu1 = (float*)ws.take((size_t)NPTS * 128 * sizeof(float));
  { Bump s = ws; gemm(cat1, NPTS, 256, PR(P_dec1pW), 256, 128, PR(P_dec1pB), xu1, 0, s); }
  { Bump s = ws; bimamba(xu1, NPTS, 128, c128, P_DEC1, s); }

  // ---- Output norm + restore original point order ----------------------------
  float* lnout = (float*)ws.take((size_t)NPTS * 128 * sizeof(float));
  k_ln_affine<<<NPTS, 256, 0, stream>>>(xu1, 128, PR(P_outG), PR(P_outBeta), lnout, 1e-5f);
  k_scatter_out<<<cdiv(NPTS * 128, 256), 256, 0, stream>>>(sidx, lnout, (float*)d_out,
                                                           NPTS, 128);
}
